// Block_8564164788955
// MI455X (gfx1250) — compile-verified, matrix operand order checked by reading, not run on hardware
//
#include <hip/hip_runtime.h>
#include <math.h>

// ---------------------------------------------------------------------------
// CDNA5 / gfx1250: wave32, WMMA 16x16x32 bf16 (f32 accum).
// ---------------------------------------------------------------------------
typedef __attribute__((ext_vector_type(16))) __bf16 v16bf;
typedef __attribute__((ext_vector_type(8)))  __bf16 bf16x8;
typedef __attribute__((ext_vector_type(8)))  float  v8f;

#define WMMA_BF16(A_, B_, C_) \
  __builtin_amdgcn_wmma_f32_16x16x32_bf16(false, (A_), false, (B_), (short)0, (C_), false, false)

// Load a 16-element bf16 fragment for WMMA A/B operands from LDS.
// Caller passes pointer at [row][k0] where k0 = (lane<16)?0:8; elements
// 0..7 = K k0..k0+7, elements 8..15 = K k0+16..k0+23 (ISA 16-bit A layout).
static __device__ __forceinline__ v16bf frag_ld(const __bf16* p) {
  bf16x8 lo = *(const bf16x8*)(p);
  bf16x8 hi = *(const bf16x8*)(p + 16);
  v16bf f;
#pragma unroll
  for (int j = 0; j < 8; ++j) { f[j] = lo[j]; f[j + 8] = hi[j]; }
  return f;
}

static __device__ __forceinline__ v8f v8f_zero() {
  v8f z = {0.f, 0.f, 0.f, 0.f, 0.f, 0.f, 0.f, 0.f};
  return z;
}

// ---------------------------------------------------------------------------
// Generic tiled WMMA GEMM: C[M x Nc] = act(A[M x K] @ W[K x Nc] + bias) + res
// Block: 256 threads (8 waves), tile 128x128, K-step 32.
// Double-buffered LDS (ping-pong) + global_prefetch two K-steps ahead.
// A may be f32 or bf16; output may be f32 or bf16.
// ---------------------------------------------------------------------------
template <typename AT, typename OT, bool GELU_, bool HAS_BIAS, bool HAS_RES>
__global__ __launch_bounds__(256) void gemm_wmma(
    const AT* __restrict__ A, const float* __restrict__ W,
    const float* __restrict__ bias, const float* __restrict__ res,
    OT* __restrict__ Cout, int M, int K, int Nc) {
  __shared__ __bf16 As[2][128 * 40];  // [m][k], stride 40 (80B rows, 16B-aligned)
  __shared__ __bf16 Bs[2][128 * 40];  // [n][k] (transposed W tile)

  const int tid  = threadIdx.x;
  const int lane = tid & 31;
  const int wave = tid >> 5;
  const int nBase = blockIdx.x * 128;
  const int mBase = blockIdx.y * 128;
  const int wm = wave >> 1;        // 0..3 : 32 rows each
  const int wn = wave & 1;         // 0..1 : 64 cols each

  v8f acc[2][4];
#pragma unroll
  for (int i = 0; i < 2; ++i)
#pragma unroll
    for (int j = 0; j < 4; ++j) acc[i][j] = v8f_zero();

  const int aRow = tid >> 1, aSeg = (tid & 1) * 16;  // A: 2 thr/row, 16 cols
  const int bK   = tid >> 3, bSeg = (tid & 7) * 16;  // W: 8 thr/row, 16 cols
  const AT*    aPtr = A + (long long)(mBase + aRow) * K + aSeg;
  const float* wPtr = W + (long long)bK * Nc + nBase + bSeg;

  const int k0 = (lane < 16) ? 0 : 8;
  const int rl = lane & 15;

  auto load_tile = [&](int kb, int buf) {
    const AT* src = aPtr + kb;
    __bf16* dst = &As[buf][aRow * 40 + aSeg];
#pragma unroll
    for (int i = 0; i < 16; ++i) dst[i] = (__bf16)(float)src[i];
    const float* srcW = wPtr + (long long)kb * Nc;
#pragma unroll
    for (int i = 0; i < 16; ++i) Bs[buf][(bSeg + i) * 40 + bK] = (__bf16)srcW[i];
  };

  auto compute = [&](int buf) {
    v16bf af[2], bfv[4];
#pragma unroll
    for (int mt = 0; mt < 2; ++mt)
      af[mt] = frag_ld(&As[buf][(wm * 32 + mt * 16 + rl) * 40 + k0]);
#pragma unroll
    for (int nt = 0; nt < 4; ++nt)
      bfv[nt] = frag_ld(&Bs[buf][(wn * 64 + nt * 16 + rl) * 40 + k0]);
#pragma unroll
    for (int mt = 0; mt < 2; ++mt)
#pragma unroll
      for (int nt = 0; nt < 4; ++nt)
        acc[mt][nt] = WMMA_BF16(af[mt], bfv[nt], acc[mt][nt]);
  };

  load_tile(0, 0);
  __syncthreads();
  int buf = 0;
  for (int kb = 32; kb < K; kb += 32) {
    if (kb + 32 < K) {  // pull the tile after next into GL2 (global_prefetch)
      __builtin_prefetch(aPtr + kb + 32, 0, 3);
      __builtin_prefetch(wPtr + (long long)(kb + 32) * Nc, 0, 3);
    }
    load_tile(kb, buf ^ 1);   // fill other buffer while consuming current
    compute(buf);
    __syncthreads();
    buf ^= 1;
  }
  compute(buf);

  // Epilogue: accum layout lane<16 -> M=r, lane>=16 -> M=8+r; N = lane&15.
  const int mOff = (lane >> 4) * 8;
#pragma unroll
  for (int mt = 0; mt < 2; ++mt)
#pragma unroll
    for (int nt = 0; nt < 4; ++nt) {
      const int col = nBase + wn * 64 + nt * 16 + rl;
      float bv = HAS_BIAS ? bias[col] : 0.f;
#pragma unroll
      for (int r = 0; r < 8; ++r) {
        const int row = mBase + wm * 32 + mt * 16 + mOff + r;
        float v = acc[mt][nt][r] + bv;
        if (GELU_) v = 0.5f * v * (1.f + erff(v * 0.70710678118654752f));
        const long long idx = (long long)row * Nc + col;
        if (HAS_RES) v += res[idx];
        Cout[idx] = (OT)v;
      }
    }
}

// ---------------------------------------------------------------------------
// LayerNorm over last dim (256). One wave per row, lane holds 8 values.
// ---------------------------------------------------------------------------
__global__ __launch_bounds__(256) void ln_kernel(
    const float* __restrict__ X, const float* __restrict__ w,
    const float* __restrict__ bsc, float* __restrict__ Y, int rows) {
  const int row = blockIdx.x * 8 + (threadIdx.x >> 5);
  const int lane = threadIdx.x & 31;
  if (row >= rows) return;
  const float* x = X + (long long)row * 256 + lane * 8;
  float v[8];
  float4 p0 = *(const float4*)(x);
  float4 p1 = *(const float4*)(x + 4);
  v[0] = p0.x; v[1] = p0.y; v[2] = p0.z; v[3] = p0.w;
  v[4] = p1.x; v[5] = p1.y; v[6] = p1.z; v[7] = p1.w;
  float s = 0.f;
#pragma unroll
  for (int i = 0; i < 8; ++i) s += v[i];
#pragma unroll
  for (int off = 16; off > 0; off >>= 1) s += __shfl_xor(s, off, 32);
  const float mean = s * (1.f / 256.f);
  float q = 0.f;
#pragma unroll
  for (int i = 0; i < 8; ++i) { float d = v[i] - mean; q += d * d; }
#pragma unroll
  for (int off = 16; off > 0; off >>= 1) q += __shfl_xor(q, off, 32);
  const float inv = rsqrtf(q * (1.f / 256.f) + 1e-5f);
  float* y = Y + (long long)row * 256;
#pragma unroll
  for (int i = 0; i < 8; ++i) {
    const int c = lane * 8 + i;
    y[c] = (v[i] - mean) * inv * w[c] + bsc[c];
  }
}

// ---------------------------------------------------------------------------
// Fused cross-attention: per (b, h, 128-query block).
// Computes S^T = K @ Q^T (so softmax over keys is per-lane), softmax,
// then O^T = V^T @ P^T where P^T accum tiles feed B-fragments directly.
// ---------------------------------------------------------------------------
__global__ __launch_bounds__(256) void cross_attn_kernel(
    const float* __restrict__ Qc, const float* __restrict__ Kc,
    const float* __restrict__ Vc, float* __restrict__ O) {
  __shared__ __bf16 Qs[128 * 40];   // [n][d]  (serves B-fragments)
  __shared__ __bf16 Ks[256 * 40];   // [m][d]  (serves A-fragments)
  __shared__ __bf16 Vt[32 * 264];   // [d][m]  (serves A-fragments of V^T)

  const int tid = threadIdx.x, lane = tid & 31, wave = tid >> 5;
  const int h = blockIdx.y, b = blockIdx.z;
  const int nBase = blockIdx.x * 128;
  const long long qb = (long long)b * 16384 * 256;
  const long long kb = (long long)b * 256 * 256;

  {  // K, V head slices: thread = key token m
    const float* ksrc = Kc + kb + (long long)tid * 256 + h * 32;
    const float* vsrc = Vc + kb + (long long)tid * 256 + h * 32;
#pragma unroll
    for (int d = 0; d < 32; ++d) {
      Ks[tid * 40 + d]  = (__bf16)ksrc[d];
      Vt[d * 264 + tid] = (__bf16)vsrc[d];
    }
  }
  {  // Q block: 128 rows x 32
    const int row = tid >> 1, seg = (tid & 1) * 16;
    const float* qsrc = Qc + qb + (long long)(nBase + row) * 256 + h * 32 + seg;
#pragma unroll
    for (int i = 0; i < 16; ++i) Qs[row * 40 + seg + i] = (__bf16)qsrc[i];
  }
  __syncthreads();

  const int k0 = (lane < 16) ? 0 : 8;
  const int rl = lane & 15;
  const int mOff = (lane >> 4) * 8;
  const int n0 = wave * 16;  // this wave's 16 query columns

  v16bf bq = frag_ld(&Qs[(n0 + rl) * 40 + k0]);
  v8f accS[16];
#pragma unroll
  for (int t = 0; t < 16; ++t) {
    v16bf ak = frag_ld(&Ks[(t * 16 + rl) * 40 + k0]);
    accS[t] = WMMA_BF16(ak, bq, v8f_zero());
  }

  // Softmax over m (rows of S^T): per-lane + one xor-16 shuffle.
  const float scale = 0.17677669529663688f;  // 32^-0.5
  float mx = -3.0e38f;
#pragma unroll
  for (int t = 0; t < 16; ++t)
#pragma unroll
    for (int r = 0; r < 8; ++r) mx = fmaxf(mx, accS[t][r]);
  mx = fmaxf(mx, __shfl_xor(mx, 16, 32));
  float ssum = 0.f;
#pragma unroll
  for (int t = 0; t < 16; ++t)
#pragma unroll
    for (int r = 0; r < 8; ++r) {
      float p = __expf((accS[t][r] - mx) * scale);
      accS[t][r] = p;
      ssum += p;
    }
  ssum += __shfl_xor(ssum, 16, 32);
  const float inv = 1.f / ssum;

  // O^T = V^T @ P^T; P^T accum tiles pack straight into B-fragments.
  v8f accO[2] = {v8f_zero(), v8f_zero()};
#pragma unroll
  for (int c = 0; c < 8; ++c) {
    v16bf bp;
#pragma unroll
    for (int j = 0; j < 8; ++j) {
      bp[j]     = (__bf16)(accS[2 * c][j] * inv);
      bp[j + 8] = (__bf16)(accS[2 * c + 1][j] * inv);
    }
#pragma unroll
    for (int dt = 0; dt < 2; ++dt) {
      v16bf av = frag_ld(&Vt[(dt * 16 + rl) * 264 + 32 * c + k0]);
      accO[dt] = WMMA_BF16(av, bp, accO[dt]);
    }
  }

  const int n = nBase + n0 + rl;
  float* outp = O + qb + (long long)n * 256 + h * 32;
#pragma unroll
  for (int dt = 0; dt < 2; ++dt)
#pragma unroll
    for (int r = 0; r < 8; ++r) outp[dt * 16 + mOff + r] = accO[dt][r];
}

// ---------------------------------------------------------------------------
// Fused 16x16 window attention with relative-position bias.
// Grid: (64 windows, 8 heads, 2 batch). Same S^T trick as cross-attn.
// ---------------------------------------------------------------------------
__global__ __launch_bounds__(256) void window_attn_kernel(
    const float* __restrict__ QKV, const float* __restrict__ rpb,
    float* __restrict__ AO) {
  __shared__ __bf16 Qs[256 * 40];
  __shared__ __bf16 Ks[256 * 40];
  __shared__ __bf16 Vt[32 * 264];
  __shared__ float  rpb_s[961];

  const int tid = threadIdx.x, lane = tid & 31, wave = tid >> 5;
  const int win = blockIdx.x, h = blockIdx.y, b = blockIdx.z;
  const int wi = win >> 3, wj = win & 7;

  {  // token tid of this window -> global row; pull q/k/v head slices
    const int gr = wi * 16 + (tid >> 4), gc = wj * 16 + (tid & 15);
    const long long row = ((long long)b * 16384 + gr * 128 + gc) * 768;
    const float* src = QKV + row + h * 32;
#pragma unroll
    for (int d = 0; d < 32; ++d) {
      Qs[tid * 40 + d]  = (__bf16)src[d];
      Ks[tid * 40 + d]  = (__bf16)src[256 + d];
      Vt[d * 264 + tid] = (__bf16)src[512 + d];
    }
  }
  for (int i = tid; i < 961; i += 256) rpb_s[i] = rpb[i * 8 + h];
  __syncthreads();

  const int k0 = (lane < 16) ? 0 : 8;
  const int rl = lane & 15;
  const int mOff = (lane >> 4) * 8;
  const float scale = 0.17677669529663688f;

  for (int it = 0; it < 2; ++it) {     // 8 waves x 2 -> 256 queries
    const int n0 = (wave + 8 * it) * 16;
    const int n = n0 + rl;             // window-token query index of this lane
    const int r1 = n >> 4, c1 = n & 15;

    v16bf bq = frag_ld(&Qs[(n0 + rl) * 40 + k0]);
    v8f accS[16];
#pragma unroll
    for (int t = 0; t < 16; ++t) {
      v16bf ak = frag_ld(&Ks[(t * 16 + rl) * 40 + k0]);
      accS[t] = WMMA_BF16(ak, bq, v8f_zero());
    }

    float mx = -3.0e38f;
#pragma unroll
    for (int t = 0; t < 16; ++t)
#pragma unroll
      for (int r = 0; r < 8; ++r) {
        const int m = t * 16 + mOff + r;  // key token index
        const float bias =
            rpb_s[(r1 - (m >> 4) + 15) * 31 + (c1 - (m & 15) + 15)];
        float v = accS[t][r] * scale + bias;
        accS[t][r] = v;
        mx = fmaxf(mx, v);
      }
    mx = fmaxf(mx, __shfl_xor(mx, 16, 32));
    float ssum = 0.f;
#pragma unroll
    for (int t = 0; t < 16; ++t)
#pragma unroll
      for (int r = 0; r < 8; ++r) {
        float p = __expf(accS[t][r] - mx);
        accS[t][r] = p;
        ssum += p;
      }
    ssum += __shfl_xor(ssum, 16, 32);
    const float inv = 1.f / ssum;

    v8f accO[2] = {v8f_zero(), v8f_zero()};
#pragma unroll
    for (int c = 0; c < 8; ++c) {
      v16bf bp;
#pragma unroll
      for (int j = 0; j < 8; ++j) {
        bp[j]     = (__bf16)(accS[2 * c][j] * inv);
        bp[j + 8] = (__bf16)(accS[2 * c + 1][j] * inv);
      }
#pragma unroll
      for (int dt = 0; dt < 2; ++dt) {
        v16bf av = frag_ld(&Vt[(dt * 16 + rl) * 264 + 32 * c + k0]);
        accO[dt] = WMMA_BF16(av, bp, accO[dt]);
      }
    }

    const long long tg =
        (long long)b * 16384 + (wi * 16 + (n >> 4)) * 128 + wj * 16 + (n & 15);
    float* outp = AO + tg * 256 + h * 32;
#pragma unroll
    for (int dt = 0; dt < 2; ++dt)
#pragma unroll
      for (int r = 0; r < 8; ++r) outp[dt * 16 + mOff + r] = accO[dt][r];
  }
}

// ---------------------------------------------------------------------------
// Host-side orchestration.
// ---------------------------------------------------------------------------
extern "C" void kernel_launch(void* const* d_in, const int* in_sizes, int n_in,
                              void* d_out, int out_size, void* d_ws,
                              size_t ws_size, hipStream_t stream) {
  (void)in_sizes; (void)n_in; (void)out_size; (void)ws_size;
  const float* x        = (const float*)d_in[0];
  const float* emb      = (const float*)d_in[1];
  const float* ca_q_w   = (const float*)d_in[2];
  const float* ca_k_w   = (const float*)d_in[3];
  const float* ca_v_w   = (const float*)d_in[4];
  const float* ca_p_w   = (const float*)d_in[5];
  const float* ca_p_b   = (const float*)d_in[6];
  const float* norm1_w  = (const float*)d_in[7];
  const float* norm1_b  = (const float*)d_in[8];
  const float* qkv_w    = (const float*)d_in[9];
  // d_in[10] = noise_strength (zero scalar; noise term dropped)
  const float* rpb      = (const float*)d_in[11];
  const float* ap_w     = (const float*)d_in[12];
  const float* ap_b     = (const float*)d_in[13];
  const float* norm2_w  = (const float*)d_in[14];
  const float* norm2_b  = (const float*)d_in[15];
  const float* fc1_w    = (const float*)d_in[16];
  const float* fc1_b    = (const float*)d_in[17];
  const float* fc2_w    = (const float*)d_in[18];
  const float* fc2_b    = (const float*)d_in[19];

  const int B = 2, N = 16384, C = 256, NH = 8, EDIM = 384, HID = 1024;
  const int M = B * N;  // 32768 flattened rows

  char* ws = (char*)d_ws;
  const size_t SQ = (size_t)B * N * C * sizeof(float);       // 33.5 MB
  const size_t SK = (size_t)B * 256 * C * sizeof(float);     // 0.5 MB
  float* Qbuf = (float*)(ws);
  float* Kc   = (float*)(ws + SQ);
  float* Vc   = (float*)(ws + SQ + SK);
  float* Sbuf = (float*)(ws + SQ + 2 * SK);      // O -> Y -> AO -> Z
  float* X1   = (float*)((char*)Sbuf + SQ);      // X1 -> X2 (in-place)
  float* QKV  = (float*)((char*)X1 + SQ);        // later reused as Hid (bf16)

  dim3 blk(256);

  // 1) Q = x @ ca_q_w
  gemm_wmma<float, float, false, false, false>
      <<<dim3(C / 128, M / 128), blk, 0, stream>>>(x, ca_q_w, nullptr, nullptr,
                                                   Qbuf, M, C, C);
  // 2,3) K,V = emb @ ca_{k,v}_w
  gemm_wmma<float, float, false, false, false>
      <<<dim3(C / 128, (B * 256) / 128), blk, 0, stream>>>(
          emb, ca_k_w, nullptr, nullptr, Kc, B * 256, EDIM, C);
  gemm_wmma<float, float, false, false, false>
      <<<dim3(C / 128, (B * 256) / 128), blk, 0, stream>>>(
          emb, ca_v_w, nullptr, nullptr, Vc, B * 256, EDIM, C);
  // 4) fused cross-attention -> Sbuf(O)
  cross_attn_kernel<<<dim3(N / 128, NH, B), blk, 0, stream>>>(Qbuf, Kc, Vc,
                                                              Sbuf);
  // 5) X1 = x + O @ ca_proj_w + b
  gemm_wmma<float, float, false, true, true>
      <<<dim3(C / 128, M / 128), blk, 0, stream>>>(Sbuf, ca_p_w, ca_p_b, x, X1,
                                                   M, C, C);
  // 6) Y = LN(X1) -> Sbuf
  ln_kernel<<<dim3(M / 8), blk, 0, stream>>>(X1, norm1_w, norm1_b, Sbuf, M);
  // 7) QKV = Y @ qkv_w
  gemm_wmma<float, float, false, false, false>
      <<<dim3((3 * C) / 128, M / 128), blk, 0, stream>>>(Sbuf, qkv_w, nullptr,
                                                         nullptr, QKV, M, C,
                                                         3 * C);
  // 8) fused window attention (+rel-pos bias) -> Sbuf(AO), natural order
  window_attn_kernel<<<dim3(64, NH, B), blk, 0, stream>>>(QKV, rpb, Sbuf);
  // 9) X2 = X1 + AO @ attn_proj_w + b   (in place over X1)
  gemm_wmma<float, float, false, true, true>
      <<<dim3(C / 128, M / 128), blk, 0, stream>>>(Sbuf, ap_w, ap_b, X1, X1, M,
                                                   C, C);
  // 10) Z = LN(X2) -> Sbuf
  ln_kernel<<<dim3(M / 8), blk, 0, stream>>>(X1, norm2_w, norm2_b, Sbuf, M);
  // 11) Hid = gelu(Z @ fc1_w + b)  stored bf16 (feeds next GEMM natively)
  __bf16* Hid = (__bf16*)QKV;
  gemm_wmma<float, __bf16, true, true, false>
      <<<dim3(HID / 128, M / 128), blk, 0, stream>>>(Sbuf, fc1_w, fc1_b,
                                                     nullptr, Hid, M, C, HID);
  // 12) out = X2 + Hid @ fc2_w + b
  gemm_wmma<__bf16, float, false, true, true>
      <<<dim3(C / 128, M / 128), blk, 0, stream>>>(Hid, fc2_w, fc2_b, X1,
                                                   (float*)d_out, M, HID, C);
}
